// HESModel_86225763435428
// MI455X (gfx1250) — compile-verified, matrix operand order checked by reading, not run on hardware
//
#include <hip/hip_runtime.h>
#include <hip/hip_bf16.h>

// ---------------------------------------------------------------------------
// Model dims
// ---------------------------------------------------------------------------
#define NGN   100000
#define EGN   800000
#define NSCN  20000
#define ESCN  80000
#define BATCH 2048
#define ATOMF 128
#define SCF   129
#define HID   256
#define EMB   256
#define NPROP 8

typedef __attribute__((ext_vector_type(16))) _Float16 v16h;
typedef __attribute__((ext_vector_type(8)))  _Float16 v8h;
typedef __attribute__((ext_vector_type(8)))  float    v8f;

// ---------------------------------------------------------------------------
// WMMA GEMM: Y[M,N] = act(A16[M,Kpad] @ Bt16[N,Kpad]^T + bias)
// One wave per block; wave computes a 16-row M tile x (NT*16) columns,
// N tiled across blockIdx.y. NT=8 keeps accumulators (64 VGPRs) + pipelined
// A/B tiles well under the register budget (no scratch spills), and
// __launch_bounds__(32,1) grants the full per-wave VGPR file.
// CDNA5 16-bit A layout: lane l holds row (l&15); elements 0..7 = K
// (l>>4)*8 .. +7, elements 8..15 = K 16+(l>>4)*8 .. +7 within each 32-K step.
// B mirrored with column (l&15) per lane. C/D: lane holds col (l&15),
// VGPR r holds row r + 8*(l>>4).
// ---------------------------------------------------------------------------
template <int NT>
__global__ void __launch_bounds__(32, 1)
gemm_wmma(const _Float16* __restrict__ A,
          const _Float16* __restrict__ Bt,
          const float* __restrict__ bias,
          float* __restrict__ Yf,
          _Float16* __restrict__ Y16,
          int Kpad, int ldY, int ldY16, int Nout, int relu) {
  const int lane  = threadIdx.x;
  const int mt    = blockIdx.x;
  const int nbase = blockIdx.y * (NT * 16);
  const int half  = lane >> 4;
  const int l15   = lane & 15;
  const _Float16* arow = A + (size_t)(mt * 16 + l15) * Kpad + half * 8;
  const _Float16* btp  = Bt + (size_t)(nbase + l15) * Kpad + half * 8;
  const size_t bstep = (size_t)16 * Kpad;

  v8f acc[NT];
#pragma unroll
  for (int i = 0; i < NT; ++i) acc[i] = (v8f){0.f,0.f,0.f,0.f,0.f,0.f,0.f,0.f};

  for (int k0 = 0; k0 < Kpad; k0 += 32) {
    v8h alo = *(const v8h*)(arow + k0);
    v8h ahi = *(const v8h*)(arow + k0 + 16);
    v16h a;
#pragma unroll
    for (int i = 0; i < 8; ++i) { a[i] = alo[i]; a[i + 8] = ahi[i]; }
    const _Float16* brow = btp + k0;
#pragma unroll
    for (int nt = 0; nt < NT; ++nt) {
      v8h blo = *(const v8h*)(brow);
      v8h bhi = *(const v8h*)(brow + 16);
      v16h b;
#pragma unroll
      for (int i = 0; i < 8; ++i) { b[i] = blo[i]; b[i + 8] = bhi[i]; }
      acc[nt] = __builtin_amdgcn_wmma_f32_16x16x32_f16(
          false, a, false, b, (short)0, acc[nt], false, false);
      brow += bstep;
    }
  }

#pragma unroll
  for (int nt = 0; nt < NT; ++nt) {
    int n = nbase + nt * 16 + l15;
    if (n < Nout) {
      float bv = bias ? bias[n] : 0.f;
#pragma unroll
      for (int r = 0; r < 8; ++r) {
        int m = mt * 16 + r + half * 8;
        float v = acc[nt][r] + bv;
        if (relu) v = fmaxf(v, 0.f);
        if (Yf)  Yf[(size_t)m * ldY + n] = v;
        if (Y16) Y16[(size_t)m * ldY16 + n] = (_Float16)v;
      }
    }
  }
}

// ---------------------------------------------------------------------------
// Elementwise / helper kernels
// ---------------------------------------------------------------------------
__global__ void fill_zero(float* p, size_t n) {
  size_t i = (size_t)blockIdx.x * 256 + threadIdx.x;
  if (i < n) p[i] = 0.f;
}

// f32 [n,kin] -> f16 [n,kpad] (zero padded)
__global__ void convert_act(const float* __restrict__ X, _Float16* __restrict__ A,
                            size_t n, int kin, int kpad) {
  size_t i = (size_t)blockIdx.x * 256 + threadIdx.x;
  size_t total = n * (size_t)kpad;
  if (i >= total) return;
  size_t r = i / kpad;
  int c = (int)(i % kpad);
  A[i] = (c < kin) ? (_Float16)X[r * (size_t)kin + c] : (_Float16)0.f;
}

// W f32 [kin,nin] -> Wt f16 [npad,kpad] (transposed, zero padded)
__global__ void convert_w(const float* __restrict__ W, _Float16* __restrict__ Wt,
                          int kin, int kpad, int nin, int npad) {
  size_t i = (size_t)blockIdx.x * 256 + threadIdx.x;
  size_t total = (size_t)npad * kpad;
  if (i >= total) return;
  int nr = (int)(i / kpad);
  int c = (int)(i % kpad);
  Wt[i] = (nr < nin && c < kin) ? (_Float16)W[(size_t)c * nin + nr] : (_Float16)0.f;
}

// z = (1+eps)*h
__global__ void zinit(const float* __restrict__ h, const float* __restrict__ eps,
                      float* __restrict__ z, size_t total) {
  size_t i = (size_t)blockIdx.x * 256 + threadIdx.x;
  if (i >= total) return;
  z[i] = (1.0f + eps[0]) * h[i];
}

// one block per edge: z[dst] += h[src]
__global__ void edge_scatter(const float* __restrict__ h, const int* __restrict__ src,
                             const int* __restrict__ dst, float* __restrict__ z) {
  int e = blockIdx.x;
  int c = threadIdx.x;
  int s = src[e], d = dst[e];
  atomicAdd(&z[(size_t)d * HID + c], h[(size_t)s * HID + c]);
}

// per-column sum & sumsq -> stats[0..255]=sum, stats[256..511]=sumsq
__global__ void bn_stats(const float* __restrict__ X, int n, float* __restrict__ stats) {
  int c = threadIdx.x;
  long r0 = (long)blockIdx.x * 128;
  long rend = r0 + 128; if (rend > n) rend = n;
  float s = 0.f, s2 = 0.f;
  for (long r = r0; r < rend; ++r) {
    float v = X[r * HID + c];
    s += v; s2 += v * v;
  }
  atomicAdd(&stats[c], s);
  atomicAdd(&stats[HID + c], s2);
}

// y = g*(x-m)*rsqrt(var+1e-5)+b, optional relu; writes f32 and/or f16
__global__ void bn_apply(const float* __restrict__ X, const float* __restrict__ stats,
                         const float* __restrict__ g, const float* __restrict__ bta,
                         float invn, float* __restrict__ Yf, _Float16* __restrict__ Y16,
                         size_t total, int relu) {
  size_t i = (size_t)blockIdx.x * 256 + threadIdx.x;
  if (i >= total) return;
  int c = (int)(i & (HID - 1));
  float m = stats[c] * invn;
  float var = stats[HID + c] * invn - m * m;
  float sc = g[c] * rsqrtf(var + 1e-5f);
  float v = (X[i] - m) * sc + bta[c];
  if (relu) v = fmaxf(v, 0.f);
  if (Yf)  Yf[i] = v;
  if (Y16) Y16[i] = (_Float16)v;
}

__global__ void count_rows(const int* __restrict__ batch, int n, float* __restrict__ counts) {
  int i = blockIdx.x * 256 + threadIdx.x;
  if (i < n) atomicAdd(&counts[batch[i]], 1.0f);
}

__global__ void seg_sum(const float* __restrict__ X, const int* __restrict__ batch,
                        float* __restrict__ seg) {
  int r = blockIdx.x;
  int c = threadIdx.x;
  atomicAdd(&seg[(size_t)batch[r] * HID + c], X[(size_t)r * HID + c]);
}

__global__ void seg_div(float* __restrict__ seg, const float* __restrict__ counts, size_t total) {
  size_t i = (size_t)blockIdx.x * 256 + threadIdx.x;
  if (i >= total) return;
  seg[i] /= fmaxf(counts[i >> 8], 1.0f);
}

// one wave per 256-wide row: x / max(||x||2, 1e-12)
__global__ void l2norm_rows(const float* __restrict__ X, float* __restrict__ Y) {
  int lane = threadIdx.x;
  const float* row = X + (size_t)blockIdx.x * 256;
  float v[8]; float s = 0.f;
#pragma unroll
  for (int i = 0; i < 8; ++i) { v[i] = row[lane + 32 * i]; s += v[i] * v[i]; }
#pragma unroll
  for (int m = 16; m; m >>= 1) s += __shfl_xor(s, m, 32);
  float inv = 1.0f / fmaxf(sqrtf(s), 1e-12f);
  float* yr = Y + (size_t)blockIdx.x * 256;
#pragma unroll
  for (int i = 0; i < 8; ++i) yr[lane + 32 * i] = v[i] * inv;
}

// embedding gather + padding_idx=0 + L2 normalize
__global__ void emb_l2(const float* __restrict__ table, const int* __restrict__ idx,
                       float* __restrict__ Y) {
  int lane = threadIdx.x;
  int i = blockIdx.x;
  int id = idx[i];
  float* yr = Y + (size_t)i * 256;
  if (id == 0) {
#pragma unroll
    for (int k = 0; k < 8; ++k) yr[lane + 32 * k] = 0.f;
    return;
  }
  const float* row = table + (size_t)id * 256;
  float v[8]; float s = 0.f;
#pragma unroll
  for (int k = 0; k < 8; ++k) { v[k] = row[lane + 32 * k]; s += v[k] * v[k]; }
#pragma unroll
  for (int m = 16; m; m >>= 1) s += __shfl_xor(s, m, 32);
  float inv = 1.0f / fmaxf(sqrtf(s), 1e-12f);
#pragma unroll
  for (int k = 0; k < 8; ++k) yr[lane + 32 * k] = v[k] * inv;
}

// concat 4 x [B,256] f32 -> [B,1024] f16
__global__ void concat4(const float* __restrict__ a, const float* __restrict__ b,
                        const float* __restrict__ c, const float* __restrict__ d,
                        _Float16* __restrict__ out, size_t total) {
  size_t i = (size_t)blockIdx.x * 256 + threadIdx.x;
  if (i >= total) return;
  size_t row = i >> 10;
  int col = (int)(i & 1023);
  const float* src; int cc;
  if (col < 256)      { src = a; cc = col; }
  else if (col < 512) { src = b; cc = col - 256; }
  else if (col < 768) { src = c; cc = col - 512; }
  else                { src = d; cc = col - 768; }
  out[i] = (_Float16)src[row * 256 + cc];
}

// ---------------------------------------------------------------------------
// Host orchestration
// ---------------------------------------------------------------------------
static inline unsigned gr(size_t total) { return (unsigned)((total + 255) / 256); }

extern "C" void kernel_launch(void* const* d_in, const int* in_sizes, int n_in,
                              void* d_out, int out_size, void* d_ws, size_t ws_size,
                              hipStream_t stream) {
  (void)in_sizes; (void)n_in; (void)out_size; (void)ws_size;

  // ---- inputs (JAX pytree order: dict keys sorted alphabetically) ----
  const float* x_g       = (const float*)d_in[0];
  const int*   ei_g      = (const int*)d_in[1];
  const float* x_sc      = (const float*)d_in[2];
  const int*   ei_sc     = (const int*)d_in[3];
  const int*   motif_idx = (const int*)d_in[4];
  const int*   shape_idx = (const int*)d_in[5];
  const int*   batch_g   = (const int*)d_in[6];
  const int*   batch_sc  = (const int*)d_in[7];
  const int PG = 8;   // params_g: 33 leaves
  const int PS = 41;  // params_sc: 33 leaves
  const float* motif_table = (const float*)d_in[74];
  const float* shape_table = (const float*)d_in[75];
  const float* h_b1 = (const float*)d_in[76];
  const float* h_b2 = (const float*)d_in[77];
  const float* h_b3 = (const float*)d_in[78];
  const float* h_w1 = (const float*)d_in[79];
  const float* h_w2 = (const float*)d_in[80];
  const float* h_w3 = (const float*)d_in[81];

  // ---- outputs (flat concat, f32) ----
  float* out       = (float*)d_out;
  float* o_gmole   = out;                                // [2048,256]
  float* o_gfrag   = o_gmole + (size_t)BATCH * EMB;      // [100000,256]
  float* o_scmole  = o_gfrag + (size_t)NGN * EMB;        // [2048,256]
  float* o_scshape = o_scmole + (size_t)BATCH * EMB;     // [20000,256]
  float* o_motif   = o_scshape + (size_t)NSCN * EMB;     // [20000,256]
  float* o_shape   = o_motif + (size_t)NSCN * EMB;       // [20000,256]
  float* o_prop    = o_shape + (size_t)NSCN * EMB;       // [2048,8]

  // ---- workspace carve ----
  char* wsb = (char*)d_ws;
  size_t off = 0;
  auto alloc = [&](size_t bytes) -> char* {
    char* p = wsb + off;
    off = (off + bytes + 255) & ~(size_t)255;
    return p;
  };
  float*     bufH   = (float*)alloc((size_t)NGN * HID * 4);
  float*     bufZ   = (float*)alloc((size_t)NGN * HID * 4);
  float*     bufY   = (float*)alloc((size_t)NGN * HID * 4);
  _Float16*  bufA16 = (_Float16*)alloc((size_t)NGN * HID * 2);
  _Float16*  wstage = (_Float16*)alloc((size_t)2 << 20);
  float*     stats  = (float*)alloc(512 * 4);
  float*     seg1   = (float*)alloc((size_t)BATCH * HID * 4); // h_glob scratch
  float*     seg2   = (float*)alloc((size_t)BATCH * HID * 4); // g frag mean
  float*     seg3   = (float*)alloc((size_t)BATCH * HID * 4); // sc shape mean
  float*     counts = (float*)alloc(BATCH * 4);
  _Float16*  hg16   = (_Float16*)alloc((size_t)BATCH * HID * 2);
  _Float16*  cat16  = (_Float16*)alloc((size_t)BATCH * 1024 * 2);
  _Float16*  h1f16  = (_Float16*)alloc((size_t)BATCH * 512 * 2);
  _Float16*  h2f16  = (_Float16*)alloc((size_t)BATCH * 256 * 2);

  // ---- one GIN encoder ----
  auto run_encoder = [&](int P, const float* X, int in_dim, int kpad_in,
                         const int* edges, int nE, const int* batch, int n,
                         float* out_frag, float* out_mole,
                         float* seg_h, float* seg_frag) {
    const float* frag_b = (const float*)d_in[P + 0];
    const float* frag_w = (const float*)d_in[P + 1];
    const float* in_b   = (const float*)d_in[P + 2];
    const float* in_w   = (const float*)d_in[P + 3];
    const float* mole_b = (const float*)d_in[P + 31];
    const float* mole_w = (const float*)d_in[P + 32];
    const size_t nh = (size_t)n * HID;
    const int mt = n / 16;
    const float invn = 1.0f / (float)n;
    const dim3 g256(mt, 2);  // NT=8 -> 2 y-tiles cover N=256

    // h = relu(X @ in_w + in_b)
    convert_act<<<gr((size_t)n * kpad_in), 256, 0, stream>>>(X, bufA16, n, in_dim, kpad_in);
    convert_w<<<gr((size_t)HID * kpad_in), 256, 0, stream>>>(in_w, wstage, in_dim, kpad_in, HID, HID);
    gemm_wmma<8><<<g256, 32, 0, stream>>>(bufA16, wstage, in_b, bufH, nullptr,
                                          kpad_in, HID, 0, HID, 1);
    for (int L = 0; L < 3; ++L) {
      int Q = P + 4 + 9 * L;
      const float* b1    = (const float*)d_in[Q + 0];
      const float* b2    = (const float*)d_in[Q + 1];
      const float* bn1_b = (const float*)d_in[Q + 2];
      const float* bn1_g = (const float*)d_in[Q + 3];
      const float* bn_b  = (const float*)d_in[Q + 4];
      const float* bn_g  = (const float*)d_in[Q + 5];
      const float* epsp  = (const float*)d_in[Q + 6];
      const float* w1    = (const float*)d_in[Q + 7];
      const float* w2    = (const float*)d_in[Q + 8];

      // z = (1+eps)*h + scatter_add(h[src] -> dst)
      zinit<<<gr(nh), 256, 0, stream>>>(bufH, epsp, bufZ, nh);
      edge_scatter<<<nE, HID, 0, stream>>>(bufH, edges, edges + nE, bufZ);
      // z @ w1 + b1
      convert_act<<<gr(nh), 256, 0, stream>>>(bufZ, bufA16, n, HID, HID);
      convert_w<<<gr((size_t)HID * HID), 256, 0, stream>>>(w1, wstage, HID, HID, HID, HID);
      gemm_wmma<8><<<g256, 32, 0, stream>>>(bufA16, wstage, b1, bufY, nullptr,
                                            HID, HID, 0, HID, 0);
      // relu(bn1(.)) -> f16 for next gemm
      fill_zero<<<gr(512), 256, 0, stream>>>(stats, 512);
      bn_stats<<<(n + 127) / 128, HID, 0, stream>>>(bufY, n, stats);
      bn_apply<<<gr(nh), 256, 0, stream>>>(bufY, stats, bn1_g, bn1_b, invn,
                                           nullptr, bufA16, nh, 1);
      // @ w2 + b2
      convert_w<<<gr((size_t)HID * HID), 256, 0, stream>>>(w2, wstage, HID, HID, HID, HID);
      gemm_wmma<8><<<g256, 32, 0, stream>>>(bufA16, wstage, b2, bufY, nullptr,
                                            HID, HID, 0, HID, 0);
      // h = relu(bn(.))  (f32 for next layer, f16 for frag/mole heads)
      fill_zero<<<gr(512), 256, 0, stream>>>(stats, 512);
      bn_stats<<<(n + 127) / 128, HID, 0, stream>>>(bufY, n, stats);
      bn_apply<<<gr(nh), 256, 0, stream>>>(bufY, stats, bn_g, bn_b, invn,
                                           bufH, bufA16, nh, 1);
    }
    // frag head: l2n(h @ frag_w + frag_b)
    convert_w<<<gr((size_t)HID * EMB), 256, 0, stream>>>(frag_w, wstage, HID, HID, EMB, EMB);
    gemm_wmma<8><<<g256, 32, 0, stream>>>(bufA16, wstage, frag_b, out_frag, nullptr,
                                          HID, EMB, 0, EMB, 0);
    l2norm_rows<<<n, 32, 0, stream>>>(out_frag, out_frag);
    // segment counts
    fill_zero<<<gr(BATCH), 256, 0, stream>>>(counts, BATCH);
    count_rows<<<gr(n), 256, 0, stream>>>(batch, n, counts);
    // h_glob = seg_mean(h) ; mole = l2n(h_glob @ mole_w + mole_b)
    fill_zero<<<gr((size_t)BATCH * HID), 256, 0, stream>>>(seg_h, (size_t)BATCH * HID);
    seg_sum<<<n, HID, 0, stream>>>(bufH, batch, seg_h);
    seg_div<<<gr((size_t)BATCH * HID), 256, 0, stream>>>(seg_h, counts, (size_t)BATCH * HID);
    convert_act<<<gr((size_t)BATCH * HID), 256, 0, stream>>>(seg_h, hg16, BATCH, HID, HID);
    convert_w<<<gr((size_t)HID * EMB), 256, 0, stream>>>(mole_w, wstage, HID, HID, EMB, EMB);
    gemm_wmma<8><<<dim3(BATCH / 16, 2), 32, 0, stream>>>(hg16, wstage, mole_b, out_mole,
                                                         nullptr, HID, EMB, 0, EMB, 0);
    l2norm_rows<<<BATCH, 32, 0, stream>>>(out_mole, out_mole);
    // frag mean = seg_mean(out_frag)
    fill_zero<<<gr((size_t)BATCH * HID), 256, 0, stream>>>(seg_frag, (size_t)BATCH * HID);
    seg_sum<<<n, HID, 0, stream>>>(out_frag, batch, seg_frag);
    seg_div<<<gr((size_t)BATCH * HID), 256, 0, stream>>>(seg_frag, counts, (size_t)BATCH * HID);
  };

  // ---- encoders ----
  run_encoder(PG, x_g, ATOMF, 128, ei_g, EGN, batch_g, NGN,
              o_gfrag, o_gmole, seg1, seg2);
  run_encoder(PS, x_sc, SCF, 160, ei_sc, ESCN, batch_sc, NSCN,
              o_scshape, o_scmole, seg1, seg3);

  // ---- embedding lookups (padding_idx=0 -> zero row -> stays zero) ----
  emb_l2<<<NSCN, 32, 0, stream>>>(motif_table, motif_idx, o_motif);
  emb_l2<<<NSCN, 32, 0, stream>>>(shape_table, shape_idx, o_shape);

  // ---- head MLP: concat [B,1024] -> 512 relu -> 256 relu -> 8 ----
  concat4<<<gr((size_t)BATCH * 1024), 256, 0, stream>>>(o_gmole, seg2, o_scmole, seg3,
                                                        cat16, (size_t)BATCH * 1024);
  convert_w<<<gr((size_t)512 * 1024), 256, 0, stream>>>(h_w1, wstage, 1024, 1024, 512, 512);
  gemm_wmma<8><<<dim3(BATCH / 16, 4), 32, 0, stream>>>(cat16, wstage, h_b1, nullptr, h1f16,
                                                       1024, 0, 512, 512, 1);
  convert_w<<<gr((size_t)256 * 512), 256, 0, stream>>>(h_w2, wstage, 512, 512, 256, 256);
  gemm_wmma<8><<<dim3(BATCH / 16, 2), 32, 0, stream>>>(h1f16, wstage, h_b2, nullptr, h2f16,
                                                       512, 0, 256, 256, 1);
  convert_w<<<gr((size_t)16 * 256), 256, 0, stream>>>(h_w3, wstage, 256, 256, NPROP, 16);
  gemm_wmma<1><<<dim3(BATCH / 16, 1), 32, 0, stream>>>(h2f16, wstage, h_b3, o_prop, nullptr,
                                                       256, NPROP, 0, NPROP, 0);
}
